// MambaBlock_Temporal_59923383714386
// MI455X (gfx1250) — matrix operand order repaired and checked by reading, not compiled, not run on hardware
//
#include <hip/hip_runtime.h>
#include <math.h>

// ---------------- constants ----------------
#define D_MODEL 512
#define D_STATE 128
#define D_CONV  4
#define D_INNER 1024
#define DT_RANK 32
#define BATCH   8
#define SEQ     128              // L = 64 + 64
#define BL      (BATCH * SEQ)    // 1024 rows
#define XDBL    (DT_RANK + 2 * D_STATE)  // 288
#define SCHUNK  16               // scan LDS staging chunk (steps)

typedef __attribute__((ext_vector_type(16))) __bf16 v16bf;
typedef __attribute__((ext_vector_type(8)))  __bf16 v8bf;
typedef __attribute__((ext_vector_type(8)))  float  v8f;

// ---------------- helpers ----------------
__device__ __forceinline__ unsigned short f2bf(float f) {
    union { float f; unsigned int u; } v; v.f = f;
    unsigned int u = v.u;
    u += 0x7FFFu + ((u >> 16) & 1u);       // round-to-nearest-even
    return (unsigned short)(u >> 16);
}
__device__ __forceinline__ float sigmoidf_(float x) {
    return 1.0f / (1.0f + __expf(-x));
}
// A fragment: two aligned 16B chunks (K kb..kb+7 and kb+16..kb+23)
__device__ __forceinline__ v16bf load_a_frag(const __bf16* p) {
    v8bf lo = *(const v8bf*)(p);
    v8bf hi = *(const v8bf*)(p + 16);
    return __builtin_shufflevector(lo, hi, 0, 1, 2, 3, 4, 5, 6, 7,
                                   8, 9, 10, 11, 12, 13, 14, 15);
}
// CDNA5 async global->LDS copy (16B per lane), tracked by ASYNCcnt.
__device__ __forceinline__ void async_g2l_b128(unsigned lds_off, const void* gaddr) {
    asm volatile("global_load_async_to_lds_b128 %0, %1, off"
                 :: "v"(lds_off), "v"(gaddr)
                 : "memory");
}
__device__ __forceinline__ void wait_async0() {
    asm volatile("s_wait_asynccnt 0x0" ::: "memory");
}

// ---------------- 1. avg-pool 2x2 + concat -> Xbf [BL, D_MODEL] ----------------
__global__ void pool_concat_kernel(const float* __restrict__ x1,
                                   const float* __restrict__ x2,
                                   unsigned short* __restrict__ Xbf) {
    int idx = blockIdx.x * blockDim.x + threadIdx.x;
    if (idx >= BL * D_MODEL) return;
    int c = idx % D_MODEL;
    int l = (idx / D_MODEL) % SEQ;
    int b = idx / (D_MODEL * SEQ);
    const float* src = (l < 64) ? x1 : x2;
    int p  = l & 63;
    int oh = p >> 3, ow = p & 7;
    int base = ((b * D_MODEL + c) * 16 + 2 * oh) * 16 + 2 * ow;
    float v = 0.25f * (src[base] + src[base + 1] + src[base + 16] + src[base + 17]);
    Xbf[idx] = f2bf(v);
}

// ---------------- 2. f32 -> bf16 convert (weights keep [N,K] layout) ----------------
__global__ void tobf16_kernel(const float* __restrict__ in,
                              unsigned short* __restrict__ out, int n) {
    int idx = blockIdx.x * blockDim.x + threadIdx.x;
    if (idx < n) out[idx] = f2bf(in[idx]);
}

// ---------------- 3. bf16 WMMA GEMM: C[M,N] = A[M,K] * Bw[N,K]^T ----------------
// One wave per 32x32 macro-tile: 2 A frags x 2 B frags -> 4 WMMAs per k-step.
__global__ void gemm_bf16_kernel(const unsigned short* __restrict__ A_,
                                 const unsigned short* __restrict__ Bw_,
                                 float* __restrict__ C,
                                 int M, int N, int K) {
    const __bf16* A  = (const __bf16*)A_;
    const __bf16* Bw = (const __bf16*)Bw_;
    int wave    = threadIdx.x >> 5;               // 4 waves / block
    int tileIdx = blockIdx.x * 4 + wave;
    int tilesN  = N >> 5;
    int tiles   = (M >> 5) * tilesN;
    if (tileIdx >= tiles) return;                 // wave-uniform; EXEC stays all-1s
    int mtile = (tileIdx / tilesN) << 5;
    int ntile = (tileIdx % tilesN) << 5;

    int lane = threadIdx.x & 31;
    int half = lane >> 4;                         // 0: lanes 0-15, 1: lanes 16-31
    int lr   = lane & 15;

    v8f acc00 = {}, acc01 = {}, acc10 = {}, acc11 = {};
    const __bf16* A0 = A  + (size_t)(mtile + lr) * K;
    const __bf16* A1 = A0 + (size_t)16 * K;
    const __bf16* B0 = Bw + (size_t)(ntile + lr) * K;
    const __bf16* B1 = B0 + (size_t)16 * K;

    for (int kk = 0; kk < K; kk += 32) {
        int kb = kk + half * 8;                   // A: K 0-7/16-23 vs 8-15/24-31
        int kr = kk + half * 8;                   // B: K 0-7/16-23 vs 8-15/24-31
        // unconditional speculative prefetch of next k-step (global_prefetch_b8)
        __builtin_prefetch(A0 + kb + 32, 0, 1);
        __builtin_prefetch(B0 + kr + 32, 0, 1);
        v16bf a0 = load_a_frag(A0 + kb);
        v16bf a1 = load_a_frag(A1 + kb);
        v16bf b0 = load_a_frag(B0 + kr);
        v16bf b1 = load_a_frag(B1 + kr);
        acc00 = __builtin_amdgcn_wmma_f32_16x16x32_bf16(false, a0, false, b0,
                                                        (short)0, acc00, false, false);
        acc01 = __builtin_amdgcn_wmma_f32_16x16x32_bf16(false, a0, false, b1,
                                                        (short)0, acc01, false, false);
        acc10 = __builtin_amdgcn_wmma_f32_16x16x32_bf16(false, a1, false, b0,
                                                        (short)0, acc10, false, false);
        acc11 = __builtin_amdgcn_wmma_f32_16x16x32_bf16(false, a1, false, b1,
                                                        (short)0, acc11, false, false);
    }
    int r0 = mtile + half * 8;                    // C/D: M = half*8 + vgpr j
#pragma unroll
    for (int j = 0; j < 8; ++j) {
        float* Crow0 = C + (size_t)(r0 + j) * N;
        float* Crow1 = C + (size_t)(r0 + 16 + j) * N;
        Crow0[ntile + lr]      = acc00[j];
        Crow0[ntile + 16 + lr] = acc01[j];
        Crow1[ntile + lr]      = acc10[j];
        Crow1[ntile + 16 + lr] = acc11[j];
    }
}

// ---------------- 4. depthwise causal conv (k=4) + SiLU ----------------
__global__ void conv_silu_kernel(const float* __restrict__ xz,
                                 const float* __restrict__ cw,
                                 const float* __restrict__ cb,
                                 float* __restrict__ u,
                                 unsigned short* __restrict__ ubf) {
    int idx = blockIdx.x * blockDim.x + threadIdx.x;
    if (idx >= BL * D_INNER) return;
    int d = idx % D_INNER;
    int l = (idx / D_INNER) % SEQ;
    int b = idx / (D_INNER * SEQ);
    float s = cb[d];
#pragma unroll
    for (int k = 0; k < D_CONV; ++k) {
        int ls = l - (D_CONV - 1) + k;
        if (ls >= 0)
            s += cw[d * D_CONV + k] * xz[((size_t)(b * SEQ + ls)) * (2 * D_INNER) + d];
    }
    float v = s * sigmoidf_(s);
    u[idx]   = v;
    ubf[idx] = f2bf(v);
}

// ---------------- 5. slice x_dbl[:, :32] -> bf16 ----------------
__global__ void dt_slice_kernel(const float* __restrict__ x_dbl,
                                unsigned short* __restrict__ out) {
    int idx = blockIdx.x * blockDim.x + threadIdx.x;
    if (idx >= BL * DT_RANK) return;
    int r = idx / DT_RANK, c = idx % DT_RANK;
    out[idx] = f2bf(x_dbl[r * XDBL + c]);
}

// ---------------- 6. dt bias + softplus (in place) ----------------
__global__ void softplus_kernel(float* __restrict__ dt,
                                const float* __restrict__ dtb) {
    int idx = blockIdx.x * blockDim.x + threadIdx.x;
    if (idx >= BL * D_INNER) return;
    int d = idx % D_INNER;
    float x = dt[idx] + dtb[d];
    dt[idx] = (x > 20.0f) ? x : log1pf(expf(x));
}

// ---------------- 7. selective scan ----------------
// 8 waves/block, one (b,d) per wave, 4 state elems per lane (N=128, wave32).
// B/C rows of x_dbl are shared by all 8 waves -> stage 16-step chunks into LDS
// with CDNA5 async global->LDS copies (ASYNCcnt), then read via ds_load.
__global__ void scan_kernel(const float* __restrict__ dt,
                            const float* __restrict__ u,
                            const float* __restrict__ x_dbl,
                            const float* __restrict__ A_log,
                            float* __restrict__ ys) {
    __shared__ float sBC[SCHUNK][2 * D_STATE];    // 16 x 256 floats = 16 KB

    int t    = threadIdx.x;
    int wid  = t >> 5;
    int lane = t & 31;
    int bd   = blockIdx.x * 8 + wid;              // 0 .. 8191 (block = same b)
    int b    = bd / D_INNER;
    int d    = bd % D_INNER;

    float A[4], h[4];
#pragma unroll
    for (int j = 0; j < 4; ++j) {
        A[j] = -expf(A_log[d * D_STATE + lane + 32 * j]);
        h[j] = 0.0f;
    }
    int bL = b * SEQ;

    for (int c0 = 0; c0 < SEQ; c0 += SCHUNK) {
        // stage rows c0..c0+15, cols 32..287 (B then C) : 1024 x 16B transfers
#pragma unroll
        for (int q = 0; q < 4; ++q) {
            int i   = t + 256 * q;                // 0..1023
            int row = i >> 6;                     // 64 x float4 per row
            int c4  = i & 63;
            const float* g = x_dbl + (size_t)(bL + c0 + row) * XDBL + DT_RANK + c4 * 4;
            unsigned lds = (unsigned)(unsigned long long)&sBC[row][c4 * 4];
            async_g2l_b128(lds, g);
        }
        wait_async0();                            // my wave's copies landed in LDS
        __syncthreads();                          // all waves' copies visible

#pragma unroll 4
        for (int l = c0; l < c0 + SCHUNK; ++l) {
            size_t row = (size_t)(bL + l);
            float dtv = dt[row * D_INNER + d];
            float uv  = u [row * D_INNER + d];
            const float* sr = sBC[l - c0];
            float du = dtv * uv;
            float acc = 0.0f;
#pragma unroll
            for (int j = 0; j < 4; ++j) {
                float Bv = sr[lane + 32 * j];
                float Cv = sr[D_STATE + lane + 32 * j];
                h[j] = __expf(dtv * A[j]) * h[j] + du * Bv;
                acc += h[j] * Cv;
            }
#pragma unroll
            for (int off = 16; off > 0; off >>= 1)   // intra-wave reduce (wave32)
                acc += __shfl_xor(acc, off, 32);
            if (lane == 0) ys[row * D_INNER + d] = acc;
        }
        __syncthreads();                          // before next chunk overwrites LDS
    }
}

// ---------------- 8. gate: (ys + u*D) * silu(z) -> bf16 ----------------
__global__ void gate_kernel(const float* __restrict__ ys,
                            const float* __restrict__ u,
                            const float* __restrict__ xz,
                            const float* __restrict__ Dskip,
                            unsigned short* __restrict__ ybf) {
    int idx = blockIdx.x * blockDim.x + threadIdx.x;
    if (idx >= BL * D_INNER) return;
    int d = idx % D_INNER;
    int r = idx / D_INNER;
    float z = xz[(size_t)r * (2 * D_INNER) + D_INNER + d];
    float y = (ys[idx] + u[idx] * Dskip[d]) * (z * sigmoidf_(z));
    ybf[idx] = f2bf(y);
}

// ---------------- 9. maxpool over L pairs -> out [B,64,C] ----------------
__global__ void maxpool_kernel(const float* __restrict__ P,
                               float* __restrict__ out) {
    int idx = blockIdx.x * blockDim.x + threadIdx.x;
    if (idx >= BATCH * 64 * D_MODEL) return;
    int c  = idx % D_MODEL;
    int l2 = (idx / D_MODEL) % 64;
    int b  = idx / (D_MODEL * 64);
    size_t p0 = (size_t)(b * SEQ + 2 * l2) * D_MODEL + c;
    out[idx] = fmaxf(P[p0], P[p0 + D_MODEL]);
}

// ---------------- launch ----------------
extern "C" void kernel_launch(void* const* d_in, const int* in_sizes, int n_in,
                              void* d_out, int out_size, void* d_ws, size_t ws_size,
                              hipStream_t stream) {
    const float* x1    = (const float*)d_in[0];
    const float* x2    = (const float*)d_in[1];
    const float* Win   = (const float*)d_in[2];   // [2048, 512]  = [N,K]
    const float* convw = (const float*)d_in[3];   // [1024, 1, 4]
    const float* convb = (const float*)d_in[4];   // [1024]
    const float* Wxp   = (const float*)d_in[5];   // [288, 1024]  = [N,K]
    const float* Wdt   = (const float*)d_in[6];   // [1024, 32]   = [N,K]
    const float* dtb   = (const float*)d_in[7];   // [1024]
    const float* Alog  = (const float*)d_in[8];   // [1024, 128]
    const float* Dskip = (const float*)d_in[9];   // [1024]
    const float* Wout  = (const float*)d_in[10];  // [512, 1024]  = [N,K]
    float* out = (float*)d_out;

    char* w = (char*)d_ws;
    size_t off = 0;
    auto alloc = [&](size_t bytes) -> void* {
        void* p = w + off;
        off = (off + bytes + 255) & ~(size_t)255;
        return p;
    };
    unsigned short* Xbf    = (unsigned short*)alloc((size_t)BL * D_MODEL * 2);
    unsigned short* WinBf  = (unsigned short*)alloc((size_t)2 * D_INNER * D_MODEL * 2);
    unsigned short* WxpBf  = (unsigned short*)alloc((size_t)XDBL * D_INNER * 2);
    unsigned short* WdtBf  = (unsigned short*)alloc((size_t)D_INNER * DT_RANK * 2);
    unsigned short* WoutBf = (unsigned short*)alloc((size_t)D_MODEL * D_INNER * 2);
    float*          xz     = (float*)         alloc((size_t)BL * 2 * D_INNER * 4);
    float*          u      = (float*)         alloc((size_t)BL * D_INNER * 4);
    unsigned short* ubf    = (unsigned short*)alloc((size_t)BL * D_INNER * 2);
    float*          x_dbl  = (float*)         alloc((size_t)BL * XDBL * 4);
    unsigned short* dtbf   = (unsigned short*)alloc((size_t)BL * DT_RANK * 2);
    float*          dtbuf  = (float*)         alloc((size_t)BL * D_INNER * 4);
    float*          ys     = (float*)         alloc((size_t)BL * D_INNER * 4);
    unsigned short* ybf    = (unsigned short*)alloc((size_t)BL * D_INNER * 2);
    float*          outpre = (float*)         alloc((size_t)BL * D_MODEL * 4);
    (void)ws_size; (void)in_sizes; (void)n_in; (void)out_size;

    const int T = 256;
    auto blocks = [](int n, int t) { return (n + t - 1) / t; };

    // 1. pool + concat -> bf16 activations
    pool_concat_kernel<<<blocks(BL * D_MODEL, T), T, 0, stream>>>(x1, x2, Xbf);

    // 2. weights f32 -> bf16 (keep native [N,K] layout == WMMA B-fragment layout)
    tobf16_kernel<<<blocks(2 * D_INNER * D_MODEL, T), T, 0, stream>>>(Win,  WinBf,  2 * D_INNER * D_MODEL);
    tobf16_kernel<<<blocks(XDBL * D_INNER, T),       T, 0, stream>>>(Wxp,  WxpBf,  XDBL * D_INNER);
    tobf16_kernel<<<blocks(D_INNER * DT_RANK, T),    T, 0, stream>>>(Wdt,  WdtBf,  D_INNER * DT_RANK);
    tobf16_kernel<<<blocks(D_MODEL * D_INNER, T),    T, 0, stream>>>(Wout, WoutBf, D_MODEL * D_INNER);

    // 3. in_proj: [1024,512] x [512,2048]
    {
        int tiles = (BL / 32) * (2 * D_INNER / 32);   // 2048
        gemm_bf16_kernel<<<tiles / 4, 128, 0, stream>>>(Xbf, WinBf, xz, BL, 2 * D_INNER, D_MODEL);
    }
    // 4. conv + silu
    conv_silu_kernel<<<blocks(BL * D_INNER, T), T, 0, stream>>>(xz, convw, convb, u, ubf);

    // 5. x_proj: [1024,1024] x [1024,288]
    {
        int tiles = (BL / 32) * (XDBL / 32);          // 288
        gemm_bf16_kernel<<<tiles / 4, 128, 0, stream>>>(ubf, WxpBf, x_dbl, BL, XDBL, D_INNER);
    }
    dt_slice_kernel<<<blocks(BL * DT_RANK, T), T, 0, stream>>>(x_dbl, dtbf);

    // 6. dt_proj: [1024,32] x [32,1024], then softplus(+bias)
    {
        int tiles = (BL / 32) * (D_INNER / 32);       // 1024
        gemm_bf16_kernel<<<tiles / 4, 128, 0, stream>>>(dtbf, WdtBf, dtbuf, BL, D_INNER, DT_RANK);
    }
    softplus_kernel<<<blocks(BL * D_INNER, T), T, 0, stream>>>(dtbuf, dtb);

    // 7. selective scan: 8192 (b,d) recurrences, 8 waves/block, LDS-staged B/C
    scan_kernel<<<(BATCH * D_INNER) / 8, 256, 0, stream>>>(dtbuf, u, x_dbl, Alog, ys);

    // 8. gate
    gate_kernel<<<blocks(BL * D_INNER, T), T, 0, stream>>>(ys, u, xz, Dskip, ybf);

    // 9. out_proj: [1024,1024] x [1024,512]
    {
        int tiles = (BL / 32) * (D_MODEL / 32);       // 512
        gemm_bf16_kernel<<<tiles / 4, 128, 0, stream>>>(ybf, WoutBf, outpre, BL, D_MODEL, D_INNER);
    }
    // 10. maxpool over L pairs
    maxpool_kernel<<<blocks(BATCH * 64 * D_MODEL, T), T, 0, stream>>>(outpre, out);
}
